// EfficientALU_L8_L9_7945689497950
// MI455X (gfx1250) — matrix operand order repaired and checked by reading, not compiled, not run on hardware
//
#include <hip/hip_runtime.h>

typedef __attribute__((ext_vector_type(16))) _Float16 v16h;
typedef __attribute__((ext_vector_type(8)))  _Float16 v8h;
typedef __attribute__((ext_vector_type(8)))  float    v8f;
typedef __attribute__((ext_vector_type(4)))  float    v4f;
typedef __attribute__((ext_vector_type(4)))  unsigned int v4u;
typedef __attribute__((ext_vector_type(8)))  int v8i;
typedef __attribute__((ext_vector_type(4)))  int v4i;

#define GE 160
#define DTOK 512
#define NTOK (8*4096)
#define TILE_TOK 64
#define NTILES (NTOK/TILE_TOK)      // 512
#define HSTRIDE 200                 // halfs per h row: 160 data + [160]=1.0 + zeros to 199
#define RELU_HI 3.402823466e38f

// packed weight sizes (halfs). Biases folded into the matrices:
//  - L2: 6 K-chunks (K=192): d==160 row carries bias1, 161..191 zero
//  - L3: only N-tile 6 (cols 96..111), 6 K-chunks, d==160 row = bias2
//  - L1 compact: k={0,1,2,3}->W cols {0,1,35,36}, k==4 -> bias0
#define L2_HALFS  (2*30720)
#define L3_HALFS  (2*3072)
#define L1_HALFS  (2*5120)
#define WS_HALFS  (L2_HALFS + L3_HALFS + L1_HALFS)   // 77824 halfs = 155648 B
#define WS_DWORDS (WS_HALFS/2)                        // 38912

// LDS byte offsets
#define LDS_WL2    0
#define LDS_WL3    122880
#define LDS_WL1    135168
#define LDS_H      155648           // 8 waves * 16 rows * 200 halfs * 2B = 51200
#define LDS_SCR    206848           // 8 waves * 2 pipes * 256 f32 = 16384
#define LDS_META   223232           // 8 waves * 8 tokens * u32 = 256
#define LDS_TOTAL  223488

// ---------------------------------------------------------------------------
// Prep: repack f32 weights (+ folded biases) into WMMA B-fragment order (f16).
// B 32x16 f16 fragment, this code's own K order (A matches; wave32): lane<16: col=lane, K=0..15 contiguous;
// lane>=16: col=lane-16, K=16..31.  B[k=d][n=o] = W[o][d].
// ---------------------------------------------------------------------------
__global__ void prep_pack(const float* __restrict__ Wa, const float* __restrict__ ba,
                          const float* __restrict__ Ws, const float* __restrict__ bs,
                          _Float16* __restrict__ wsH) {
  int i = blockIdx.x * blockDim.x + threadIdx.x;
  if (i < L2_HALFS) {               // layer 2 (W index 1), 10 N-tiles, K=192
    int p = i / 30720;  int r = i % 30720;
    int f = r >> 9;     int wrd = r & 511;
    int lane = wrd >> 4, h = wrd & 15;
    int kc = f / 10,     nt = f % 10;
    int o = nt * 16 + (lane & 15);
    int d = kc * 32 + ((lane >= 16) ? 16 : 0) + h;
    const float* W = (p ? Ws : Wa) + 1 * GE * GE;
    const float* b = (p ? bs : ba) + 1 * GE;
    float v = (d < 160) ? W[o * GE + d] : ((d == 160) ? b[o] : 0.f);
    wsH[i] = (_Float16)v;
    return;
  }
  i -= L2_HALFS;
  if (i < L3_HALFS) {               // layer 3 (W index 2), only N-tile 6, K=192
    int p = i / 3072;   int r = i % 3072;
    int kc = r >> 9;    int wrd = r & 511;
    int lane = wrd >> 4, h = wrd & 15;
    int o = 6 * 16 + (lane & 15);
    int d = kc * 32 + ((lane >= 16) ? 16 : 0) + h;
    const float* W = (p ? Ws : Wa) + 2 * GE * GE;
    const float* b = (p ? bs : ba) + 2 * GE;
    float v = (d < 160) ? W[o * GE + d] : ((d == 160) ? b[o] : 0.f);
    wsH[L2_HALFS + i] = (_Float16)v;
    return;
  }
  i -= L3_HALFS;
  if (i < L1_HALFS) {               // layer 1 (W index 0), compact K + bias at k=4
    int p = i / 5120;   int r = i % 5120;
    int nt = r >> 9;    int wrd = r & 511;
    int lane = wrd >> 4, h = wrd & 15;
    int o = nt * 16 + (lane & 15);
    int k = ((lane >= 16) ? 16 : 0) + h;
    const float* W = (p ? Ws : Wa);
    const float* b = (p ? bs : ba);
    float v = 0.f;
    if      (k == 0) v = W[o * GE + 0];
    else if (k == 1) v = W[o * GE + 1];
    else if (k == 2) v = W[o * GE + 35];
    else if (k == 3) v = W[o * GE + 36];
    else if (k == 4) v = b[o];
    wsH[L2_HALFS + L3_HALFS + i] = (_Float16)v;
  }
}

__device__ __forceinline__ _Float16 relu16(float v) {
  return (_Float16)__builtin_amdgcn_fmed3f(v, 0.0f, RELU_HI);
}

// ---------------------------------------------------------------------------
// Fused main kernel: 8 waves/WG, wave owns 8 tokens (16 GEMM rows).
// Phase-batched per layer: [A loads] -> [all WMMA chains] -> [relu/cvt stores]
// so independent WMMAs pipeline without WMMA->VALU hazard NOPs.
// ---------------------------------------------------------------------------
__global__ void __launch_bounds__(256, 1)
alu_wmma_kernel(const float* __restrict__ x, const unsigned int* __restrict__ wsU32,
                float* __restrict__ out) {
  extern __shared__ char smem[];
  const int tid  = threadIdx.x;
  const int lane = tid & 31;
  const int w    = tid >> 5;
  const int lm   = lane & 15;
  const int hiH  = (lane >= 16) ? 1 : 0;
  const int moff = hiH ? 8 : 0;

  // --- TDM async prefetch of packed weights into LDS (wave 0 issues) ---
#if __has_builtin(__builtin_amdgcn_tensor_load_to_lds) && __has_builtin(__builtin_amdgcn_s_wait_tensorcnt)
  if (w == 0) {
    unsigned long long ga = (unsigned long long)(size_t)wsU32;
    v4u g0;
    g0[0] = 1u;                                          // count=1
    g0[1] = 0u;                                          // lds_addr = 0
    g0[2] = (unsigned int)(ga & 0xffffffffu);            // global_addr[31:0]
    g0[3] = (unsigned int)((ga >> 32) & 0x01ffffffu) | (2u << 30); // addr hi, type=2
    v8i g1;
    g1[0] = (int)(2u << 16);                             // data_size = 4B
    g1[1] = (int)((WS_DWORDS & 0xffffu) << 16);          // tensor_dim0 [63:48]
    g1[2] = (int)(((unsigned)WS_DWORDS >> 16) | (1u << 16)); // dim0 hi / tensor_dim1=1
    g1[3] = (int)(((unsigned)WS_DWORDS & 0xffffu) << 16);    // tile_dim0 [127:112]
    g1[4] = 1;                                           // tile_dim1 = 1
    g1[5] = (int)WS_DWORDS;                              // tensor_dim0_stride lo
    g1[6] = 0;
    g1[7] = 0;
    v4i gz; gz[0] = 0; gz[1] = 0; gz[2] = 0; gz[3] = 0;
#if defined(__clang_major__) && __clang_major__ >= 23
    v8i gz8;
    for (int q = 0; q < 8; ++q) gz8[q] = 0;
    __builtin_amdgcn_tensor_load_to_lds(g0, g1, gz, gz, gz8, 0);
#else
    __builtin_amdgcn_tensor_load_to_lds(g0, g1, gz, gz, 0);
#endif
    __builtin_amdgcn_s_wait_tensorcnt(0);
  }
  __syncthreads();
#endif

  // --- cooperative load (semantic ground truth; hits warm lines) ---
  {
    unsigned int* dst = (unsigned int*)smem;
    for (int i = tid; i < WS_DWORDS; i += 256) dst[i] = wsU32[i];
  }

  const _Float16* wl2 = (const _Float16*)(smem + LDS_WL2);
  const _Float16* wl3 = (const _Float16*)(smem + LDS_WL3);
  const _Float16* wl1 = (const _Float16*)(smem + LDS_WL1);
  _Float16* hbase = (_Float16*)(smem + LDS_H) + w * 16 * HSTRIDE;
  float*    scrw  = (float*)(smem + LDS_SCR) + w * 2 * 256;
  unsigned int* meta = (unsigned int*)(smem + LDS_META) + w * 8;

  // h tail: col 160 = 1.0 (bias multiplier), cols 161..199 = 0, persistent.
  if (lane < 16) {
    for (int i = 0; i < 40; ++i)
      hbase[lm * HSTRIDE + 160 + i] = (_Float16)((i == 0) ? 1.0f : 0.0f);
  }
  __syncthreads();

  for (int tile = blockIdx.x; tile < NTILES; tile += gridDim.x) {
    const int t0 = tile * TILE_TOK + w * 8;

    // ---------------- decode (lanes 16..31 duplicate; contribute zeros) ----
    const int m   = lm;
    const int tok = t0 + (m >> 1);
    const int pos = m & 1;
    const float* xt = x + (size_t)tok * DTOK;
    int ia = 0, ib = 0;
    {
      const int baseA = pos ? 32 : 16;   // ALU_HI : ALU_LO
      const int baseB = pos ? 64 : 48;   // AX_CARRY_HI : AX_CARRY_LO
      for (int k = 15; k >= 0; --k) {    // first idx > 0.5, else 0
        if (xt[baseA + k] > 0.5f) ia = k;
        if (xt[baseB + k] > 0.5f) ib = k;
      }
    }
    const float opadd = xt[80 + 25];
    const float opsub = xt[80 + 26];
    if (lane < 16 && pos == 0) {
      bool mark = xt[0] > 0.5f;
      bool madd = mark && (xt[1] > 0.5f);
      bool msub = mark && !(xt[1] > 0.5f) && (xt[2] > 0.5f);
      meta[m >> 1] = ((madd || msub) ? 1u : 0u) | (madd ? 2u : 0u);
    }
    // layer-1 A fragment: K0..3 = [nibA,nibB,opadd,opsub], K4 = 1.0, rest 0
    v16h a1;
    for (int i = 0; i < 16; ++i) a1[i] = (_Float16)0.f;
    if (!hiH) {
      a1[0] = (_Float16)(float)ia; a1[1] = (_Float16)(float)ib;
      a1[2] = (_Float16)opadd;     a1[3] = (_Float16)opsub;
      a1[4] = (_Float16)1.0f;
    }

    // ---------------- both MLP pipelines via WMMA ----------------
    for (int p = 0; p < 2; ++p) {
      // ---- layer 1: 10 independent WMMAs, then one store phase ----
      {
        v8f acc[10];
        for (int nt = 0; nt < 10; ++nt) {
          v8f c = {};
          v16h bf = *(const v16h*)(wl1 + p * 5120 + nt * 512 + lane * 16);
          acc[nt] = __builtin_amdgcn_wmma_f32_16x16x32_f16(false, a1, false, bf,
                                                           (short)0, c, false, false);
        }
        for (int nt = 0; nt < 10; ++nt)
          for (int r = 0; r < 8; ++r)
            hbase[(r + moff) * HSTRIDE + nt * 16 + lm] = relu16(acc[nt][r]);
      }
      // ---- layer 2: load 6 A chunks, 10 chains x 6 WMMAs, store phase ----
      {
        v16h a2[6];
        for (int kc = 0; kc < 6; ++kc) {
          const _Float16* row = hbase + lm * HSTRIDE + kc * 32 + (hiH ? 16 : 0);
          v8h lo = *(const v8h*)row;
          v8h hi = *(const v8h*)(row + 8);
          v16h a; for (int i = 0; i < 8; ++i) { a[i] = lo[i]; a[8 + i] = hi[i]; }
          a2[kc] = a;
        }
        const _Float16* mb = wl2 + p * 30720;
        v8f acc[10];
        for (int nt = 0; nt < 10; ++nt) {
          v8f c = {};
          for (int kc = 0; kc < 6; ++kc) {
            v16h bf = *(const v16h*)(mb + (kc * 10 + nt) * 512 + lane * 16);
            c = __builtin_amdgcn_wmma_f32_16x16x32_f16(false, a2[kc], false, bf,
                                                       (short)0, c, false, false);
          }
          acc[nt] = c;
        }
        for (int nt = 0; nt < 10; ++nt)
          for (int r = 0; r < 8; ++r)
            hbase[(r + moff) * HSTRIDE + nt * 16 + lm] = relu16(acc[nt][r]);
      }
      // ---- layer 3: only N-tile 6 (cols 96..111: RESULT=100, CARRY_OUT=101) ----
      {
        const _Float16* mb = wl3 + p * 3072;
        v8f c = {};
        for (int kc = 0; kc < 6; ++kc) {
          const _Float16* row = hbase + lm * HSTRIDE + kc * 32 + (hiH ? 16 : 0);
          v8h lo = *(const v8h*)row;
          v8h hi = *(const v8h*)(row + 8);
          v16h a; for (int i = 0; i < 8; ++i) { a[i] = lo[i]; a[8 + i] = hi[i]; }
          v16h bf = *(const v16h*)(mb + kc * 512 + lane * 16);
          c = __builtin_amdgcn_wmma_f32_16x16x32_f16(false, a, false, bf,
                                                     (short)0, c, false, false);
        }
        for (int r = 0; r < 8; ++r)
          scrw[p * 256 + (r + moff) * 16 + lm] = c[r];   // 16x16 f32 tile
      }
    }

    // ---------------- streamed copy + scatter-encode (roofline part) -------
    for (int j = 0; j < 8; ++j) {
      const unsigned int mt = meta[j];
      const float activef = (mt & 1u) ? 1.0f : 0.0f;
      const int   psel    = (mt & 2u) ? 0 : 1;
      const float* scrp = scrw + psel * 256;
      const float rlo = scrp[(2 * j)     * 16 + 4]; // col 100 -> n=4 of tile 6
      const float rhi = scrp[(2 * j + 1) * 16 + 4];
      const float cyf = scrp[(2 * j + 1) * 16 + 5]; // col 101
      int reslo = (int)rintf(rlo); reslo = reslo < 0 ? 0 : (reslo > 15 ? 15 : reslo);
      int reshi = (int)rintf(rhi); reshi = reshi < 0 ? 0 : (reshi > 15 ? 15 : reshi);
      const float carry = (rintf(cyf) > 0.5f) ? activef : 0.0f;
      const int t = t0 + j;
      const int tn = t + TILE_TOK * (int)gridDim.x;     // this WG's next tile
      if (tn < NTOK)
        __builtin_prefetch(x + (size_t)tn * DTOK + lane * 16, 0, 1);
      const float* xin = x   + (size_t)t * DTOK + lane * 16;
      float*       xo  = out + (size_t)t * DTOK + lane * 16;
      for (int q = 0; q < 4; ++q) {
        v4f v = *(const v4f*)(xin + q * 4);
        for (int e = 0; e < 4; ++e) {
          const int d = lane * 16 + q * 4 + e;
          float add = 0.f;
          if (d >= 128 && d < 144 && (d - 128) == reslo) add += 2.f * activef;
          if (d >= 144 && d < 160 && (d - 144) == reshi) add += 2.f * activef;
          if (d == 160) add += carry;
          v[e] += add;
        }
        *(v4f*)(xo + q * 4) = v;
      }
    }
  }
}

extern "C" void kernel_launch(void* const* d_in, const int* in_sizes, int n_in,
                              void* d_out, int out_size, void* d_ws, size_t ws_size,
                              hipStream_t stream) {
  (void)in_sizes; (void)n_in; (void)out_size; (void)ws_size;
  const float* x  = (const float*)d_in[0];
  const float* Wa = (const float*)d_in[1];
  const float* ba = (const float*)d_in[2];
  const float* Ws = (const float*)d_in[3];
  const float* bs = (const float*)d_in[4];

  prep_pack<<<(WS_HALFS + 255) / 256, 256, 0, stream>>>(Wa, ba, Ws, bs, (_Float16*)d_ws);

  alu_wmma_kernel<<<128, 256, LDS_TOTAL, stream>>>(
      x, (const unsigned int*)d_ws, (float*)d_out);
}
